// ScaledDotProductAttention_14757507629693
// MI455X (gfx1250) — matrix operand order repaired and checked by reading, not compiled, not run on hardware
//
#include <hip/hip_runtime.h>

typedef __attribute__((ext_vector_type(16))) _Float16 v16h;
typedef __attribute__((ext_vector_type(8)))  _Float16 v8h;
typedef __attribute__((ext_vector_type(8)))  float    v8f;
typedef __attribute__((ext_vector_type(4)))  float    f32x4;
typedef unsigned int u32x4 __attribute__((ext_vector_type(4)));
typedef int          i32x8 __attribute__((ext_vector_type(8)));
typedef int          i32x4 __attribute__((ext_vector_type(4)));

static constexpr int   Bsz   = 4, H = 16, S = 2048, D = 64;
static constexpr float SCALE = 0.125f;                 // 1/sqrt(64)
static constexpr float NEGM  = -3.402823466e38f;       // finfo(f32).min
static constexpr float LOG2E = 1.4426950408889634f;

static constexpr int KSTR = 72;   // K-tile LDS stride (halves): 64 + 8 pad
static constexpr int VSTR = 40;   // V^T-tile LDS stride (halves): 32 + 8 pad

#define HAVE_TDM __has_builtin(__builtin_amdgcn_tensor_load_to_lds)

__device__ __forceinline__ v8f wmma_f16(v16h a, v16h b, v8f c) {
  return __builtin_amdgcn_wmma_f32_16x16x32_f16(false, a, false, b, (short)0, c,
                                                false, false);
}
__device__ __forceinline__ float swap16(float x) {  // lane <-> lane^16
  return __builtin_bit_cast(
      float, __builtin_amdgcn_ds_swizzle(__builtin_bit_cast(int, x), 0x401F));
}
__device__ __forceinline__ float bcast(float x, int srcLane) {
  return __builtin_bit_cast(
      float, __builtin_amdgcn_ds_bpermute(srcLane << 2,
                                          __builtin_bit_cast(int, x)));
}
__device__ __forceinline__ v16h cat(v8h lo, v8h hi) {
  return __builtin_shufflevector(lo, hi, 0, 1, 2, 3, 4, 5, 6, 7, 8, 9, 10, 11,
                                 12, 13, 14, 15);
}

// ---------------- shared per-iteration flash-attention step ----------------
struct FAState {
  v8f   o0, o1, o2, o3;
  float m_run, l_run;
};

__device__ __forceinline__ void fa_step(const _Float16* __restrict__ sk,
                                        const _Float16* __restrict__ sv,
                                        const float* __restrict__ bias_kb,
                                        int lhalf, int l15, v16h bq0, v16h bq1,
                                        FAState& Z) {
  // ---- S^T = K_tile x Q^T (A-operands: two contiguous 16B LDS runs) ----
  v8f st0 = {}, st1 = {};
  {
    const _Float16* k0 = sk + l15 * KSTR + lhalf * 8;
    const _Float16* k1 = k0 + 16 * KSTR;
    v16h a;
    a = cat(*(const v8h*)(k0), *(const v8h*)(k0 + 16));
    st0 = wmma_f16(a, bq0, st0);
    a = cat(*(const v8h*)(k0 + 32), *(const v8h*)(k0 + 48));
    st0 = wmma_f16(a, bq1, st0);
    a = cat(*(const v8h*)(k1), *(const v8h*)(k1 + 16));
    st1 = wmma_f16(a, bq0, st1);
    a = cat(*(const v8h*)(k1 + 32), *(const v8h*)(k1 + 48));
    st1 = wmma_f16(a, bq1, st1);
  }
  // ---- bias + online softmax ----
  const f32x4* bp = (const f32x4*)(bias_kb + lhalf * 8);
  f32x4 bb0 = bp[0], bb1 = bp[1], bb2 = bp[4], bb3 = bp[5];
  float mx = NEGM;
#pragma unroll
  for (int r = 0; r < 4; ++r) {
    st0[r] += bb0[r];  st0[r + 4] += bb1[r];
    st1[r] += bb2[r];  st1[r + 4] += bb3[r];
    mx = fmaxf(mx,
               fmaxf(fmaxf(st0[r], st0[r + 4]), fmaxf(st1[r], st1[r + 4])));
  }
  mx = fmaxf(mx, swap16(mx));
  const float m_new = fmaxf(Z.m_run, mx);
  const float alpha = __builtin_amdgcn_exp2f((Z.m_run - m_new) * LOG2E);
  float rs = 0.0f;
#pragma unroll
  for (int r = 0; r < 8; ++r) {
    st0[r] = __builtin_amdgcn_exp2f((st0[r] - m_new) * LOG2E);
    st1[r] = __builtin_amdgcn_exp2f((st1[r] - m_new) * LOG2E);
    rs += st0[r] + st1[r];
  }
  rs += swap16(rs);
  Z.l_run = Z.l_run * alpha + rs;
  if (__builtin_amdgcn_ballot_w32(m_new > Z.m_run)) {
#pragma unroll
    for (int r = 0; r < 8; ++r) {
      const float ar = bcast(alpha, r + lhalf * 8);
      Z.o0[r] *= ar; Z.o1[r] *= ar; Z.o2[r] *= ar; Z.o3[r] *= ar;
    }
  }
  Z.m_run = m_new;
  // ---- P already in A-operand layout ----
  v16h ap;
#pragma unroll
  for (int r = 0; r < 8; ++r) {
    ap[r]     = (_Float16)st0[r];
    ap[r + 8] = (_Float16)st1[r];
  }
  // ---- P(16x32) x V(32x64) ----
#pragma unroll
  for (int t = 0; t < 4; ++t) {
    const v8h* vp = (const v8h*)(sv + (t * 16 + l15) * VSTR + lhalf * 8);
    v16h bv = cat(vp[0], vp[2]);
    v8f acc = (t == 0) ? Z.o0 : (t == 1) ? Z.o1 : (t == 2) ? Z.o2 : Z.o3;
    acc = wmma_f16(ap, bv, acc);
    if (t == 0) Z.o0 = acc; else if (t == 1) Z.o1 = acc;
    else if (t == 2) Z.o2 = acc; else Z.o3 = acc;
  }
}

__device__ __forceinline__ void load_q(const float* __restrict__ Qb, int qb,
                                       int lhalf, int l15, v16h& bq0, v16h& bq1) {
  const f32x4* qr = (const f32x4*)(Qb + (size_t)(qb + l15) * D + lhalf * 8);
#pragma unroll
  for (int v = 0; v < 4; ++v) {
    f32x4 x0 = qr[v + (v & 2)], x1 = qr[v + (v & 2) + 8];
#pragma unroll
    for (int j = 0; j < 4; ++j) {
      bq0[4 * v + j] = (_Float16)(x0[j] * SCALE);
      bq1[4 * v + j] = (_Float16)(x1[j] * SCALE);
    }
  }
}

__device__ __forceinline__ void store_out(float* __restrict__ Ob, int qb,
                                          int lhalf, int l15, const FAState& Z) {
  const float linv = 1.0f / Z.l_run;
#pragma unroll
  for (int r = 0; r < 8; ++r) {
    const float sr = bcast(linv, r + lhalf * 8);
    float* orow = Ob + (size_t)(qb + r + lhalf * 8) * D + l15;
    orow[0]  = Z.o0[r] * sr;
    orow[16] = Z.o1[r] * sr;
    orow[32] = Z.o2[r] * sr;
    orow[48] = Z.o3[r] * sr;
  }
}

__device__ __forceinline__ void init_bias(const float* __restrict__ M, int b,
                                          int tid, float* __restrict__ s_bias) {
  const float* mp = M + (size_t)b * S + tid * 8;
  f32x4 m0 = *(const f32x4*)(mp);
  f32x4 m1 = *(const f32x4*)(mp + 4);
  f32x4 b0, b1;
#pragma unroll
  for (int j = 0; j < 4; ++j) {
    b0[j] = (1.0f - m0[j]) * NEGM;
    b1[j] = (1.0f - m1[j]) * NEGM;
  }
  *(f32x4*)(s_bias + tid * 8)     = b0;
  *(f32x4*)(s_bias + tid * 8 + 4) = b1;
}

// ===================== TDM fast path =====================
#if HAVE_TDM
// Build a 2D tensor descriptor (D#) per CDNA5 ISA ch.8 and issue the DMA.
// This toolchain exposes the 6-arg builtin:
//   (u32x4 g0, i32x8 g1, i32x4 g2, i32x4 g3, i32x8 extra, i32 cpol)
__device__ __forceinline__ void tdm_load_2d(unsigned lds_off, const void* gptr,
                                            unsigned td0, unsigned td1,
                                            unsigned tile0, unsigned tile1,
                                            unsigned stride0, unsigned pad_int,
                                            unsigned pad_amt) {
  unsigned long long ga = (unsigned long long)(uintptr_t)gptr;
  u32x4 g0 = {1u /*count=1,user*/, lds_off, (unsigned)ga,
              (unsigned)(((ga >> 32) & 0x01FFFFFFull) | 0x80000000u /*type=2*/)};
  i32x8 g1;
  g1[0] = (int)((1u << 16) /*data_size=2B*/ | (1u << 20) /*pad_enable*/ |
                (pad_int << 22) | (pad_amt << 25));
  g1[1] = (int)((td0 & 0xFFFFu) << 16);                  // tensor_dim0[15:0]
  g1[2] = (int)((td0 >> 16) | ((td1 & 0xFFFFu) << 16));  // td0 hi | td1 lo
  g1[3] = (int)((td1 >> 16) | (tile0 << 16));            // td1 hi | tile_dim0
  g1[4] = (int)tile1;                                    // tile_dim1 (dim2=0)
  g1[5] = (int)stride0;                                  // tensor_dim0_stride
  g1[6] = 0;
  g1[7] = 0;
  i32x4 gz4 = {0, 0, 0, 0};
  i32x8 gz8 = {0, 0, 0, 0, 0, 0, 0, 0};
  __builtin_amdgcn_tensor_load_to_lds(g0, g1, gz4, gz4, gz8, 0);
}

// Prepack: K -> f16 row-major; V -> f16 transposed Vt[bh][d][s].
__global__ __launch_bounds__(256) void sdpa_prepack_kernel(
    const float* __restrict__ K, const float* __restrict__ V,
    _Float16* __restrict__ Kh, _Float16* __restrict__ Vt) {
  const int tid = threadIdx.x;
  const int bh  = blockIdx.x >> 3;
  const int s0  = (blockIdx.x & 7) * 256;   // 256-row chunk
  const float* Kb  = K  + ((size_t)bh * S + s0) * D;
  const float* Vb  = V  + ((size_t)bh * S + s0) * D;
  _Float16*    KhB = Kh + ((size_t)bh * S + s0) * D;
  _Float16*    VtB = Vt + (size_t)bh * D * S + s0;
  // K: flat convert, coalesced
#pragma unroll
  for (int it = 0; it < 8; ++it) {
    const int e = it * 2048 + tid * 8;
    f32x4 x0 = *(const f32x4*)(Kb + e);
    f32x4 x1 = *(const f32x4*)(Kb + e + 4);
    v8h h;
#pragma unroll
    for (int j = 0; j < 4; ++j) {
      h[j] = (_Float16)x0[j];
      h[4 + j] = (_Float16)x1[j];
    }
    *(v8h*)(KhB + e) = h;
  }
  // V^T: column strips (coalesced strided reads, contiguous 16B writes)
  const int d  = tid & 63;
  const int sg = (tid >> 6) * 8;
#pragma unroll
  for (int it = 0; it < 8; ++it) {
    const int sl = sg + it * 32;
    const float* vp = Vb + (size_t)sl * D + d;
    v8h h;
#pragma unroll
    for (int j = 0; j < 8; ++j) h[j] = (_Float16)vp[(size_t)j * D];
    *(v8h*)(VtB + (size_t)d * S + sl) = h;
  }
}

__global__ __launch_bounds__(256) void sdpa_fa_tdm_kernel(
    const float* __restrict__ Q, const _Float16* __restrict__ Kh,
    const _Float16* __restrict__ Vt, const float* __restrict__ M,
    float* __restrict__ O) {
  __shared__ alignas(16) float    s_bias[S];
  __shared__ alignas(16) _Float16 s_k[2][32 * KSTR];
  __shared__ alignas(16) _Float16 s_v[2][64 * VSTR];

  const int tid  = threadIdx.x;
  const int lane = tid & 31;
  const int wave = tid >> 5;
  const int bh   = blockIdx.x >> 4;
  const int qch  = blockIdx.x & 15;
  const int b    = bh >> 4;
  const int qb   = qch * 128 + wave * 16;
  const int lhalf = lane >> 4, l15 = lane & 15;

  const float*    Qb  = Q  + (size_t)bh * S * D;
  const _Float16* KhB = Kh + (size_t)bh * S * D;
  const _Float16* VtB = Vt + (size_t)bh * D * S;
  float*          Ob  = O  + (size_t)bh * S * D;

  const unsigned lk0 = (unsigned)(uintptr_t)&s_k[0][0];
  const unsigned lk1 = (unsigned)(uintptr_t)&s_k[1][0];
  const unsigned lv0 = (unsigned)(uintptr_t)&s_v[0][0];
  const unsigned lv1 = (unsigned)(uintptr_t)&s_v[1][0];

  // Kick off DMA of tile 0 before doing prologue work.
  if (wave == 0) {
    tdm_load_2d(lk0, KhB, 64, S, 64, 32, 64, 4 /*32DW*/, 3 /*4DW*/);
    tdm_load_2d(lv0, VtB, S, 64, 32, 64, S, 3 /*16DW*/, 3 /*4DW*/);
  }

  init_bias(M, b, tid, s_bias);
  v16h bq0, bq1;
  load_q(Qb, qb, lhalf, l15, bq0, bq1);

  FAState Z;
  Z.o0 = {}; Z.o1 = {}; Z.o2 = {}; Z.o3 = {};
  Z.m_run = -__builtin_inff();
  Z.l_run = 0.0f;

  for (int it = 0; it < S / 32; ++it) {
    const int buf = it & 1;
    if (wave == 0) __builtin_amdgcn_s_wait_tensorcnt(0);
    __syncthreads();   // tile `it` visible; everyone done with buf^1
    if (wave == 0 && it + 1 < S / 32) {
      const int kb2 = (it + 1) * 32;
      tdm_load_2d(buf ? lk0 : lk1, KhB + (size_t)kb2 * D, 64, S, 64, 32, 64, 4, 3);
      tdm_load_2d(buf ? lv0 : lv1, VtB + kb2, S, 64, 32, 64, S, 3, 3);
    }
    fa_step(s_k[buf], s_v[buf], s_bias + it * 32, lhalf, l15, bq0, bq1, Z);
  }
  store_out(Ob, qb, lhalf, l15, Z);
}
#endif  // HAVE_TDM

// ===================== fallback: self-staging (round-2) path =====================
__global__ __launch_bounds__(256) void sdpa_fa_stage_kernel(
    const float* __restrict__ Q, const float* __restrict__ K,
    const float* __restrict__ V, const float* __restrict__ M,
    float* __restrict__ O) {
  __shared__ alignas(16) float    s_bias[S];
  __shared__ alignas(16) _Float16 s_k[32 * KSTR];
  __shared__ alignas(16) _Float16 s_vt[64 * VSTR];

  const int tid  = threadIdx.x;
  const int lane = tid & 31;
  const int wave = tid >> 5;
  const int bh   = blockIdx.x >> 4;
  const int qch  = blockIdx.x & 15;
  const int b    = bh >> 4;
  const int qb   = qch * 128 + wave * 16;
  const int lhalf = lane >> 4, l15 = lane & 15;

  const float* Qb = Q + (size_t)bh * S * D;
  const float* Kb = K + (size_t)bh * S * D;
  const float* Vb = V + (size_t)bh * S * D;
  float*       Ob = O + (size_t)bh * S * D;

  init_bias(M, b, tid, s_bias);
  v16h bq0, bq1;
  load_q(Qb, qb, lhalf, l15, bq0, bq1);

  FAState Z;
  Z.o0 = {}; Z.o1 = {}; Z.o2 = {}; Z.o3 = {};
  Z.m_run = -__builtin_inff();
  Z.l_run = 0.0f;

  const int krow = tid >> 3, kdb = (tid & 7) * 8;
  const int vd   = tid & 63, vkg = (tid >> 6) * 8;

  for (int kb = 0; kb < S; kb += 32) {
    {
      const f32x4* kp = (const f32x4*)(Kb + (size_t)(kb + krow) * D + kdb);
      f32x4 x0 = kp[0], x1 = kp[1];
      v8h h;
#pragma unroll
      for (int j = 0; j < 4; ++j) {
        h[j] = (_Float16)x0[j];
        h[4 + j] = (_Float16)x1[j];
      }
      *(v8h*)(s_k + krow * KSTR + kdb) = h;
    }
    {
      const float* vp = Vb + (size_t)(kb + vkg) * D + vd;
      v8h h;
#pragma unroll
      for (int j = 0; j < 8; ++j) h[j] = (_Float16)vp[(size_t)j * D];
      *(v8h*)(s_vt + vd * VSTR + vkg) = h;
    }
    __syncthreads();
    fa_step(s_k, s_vt, s_bias + kb, lhalf, l15, bq0, bq1, Z);
    __syncthreads();
  }
  store_out(Ob, qb, lhalf, l15, Z);
}

extern "C" void kernel_launch(void* const* d_in, const int* in_sizes, int n_in,
                              void* d_out, int out_size, void* d_ws,
                              size_t ws_size, hipStream_t stream) {
  const float* Q = (const float*)d_in[0];
  const float* K = (const float*)d_in[1];
  const float* V = (const float*)d_in[2];
  const float* M = (const float*)d_in[3];
  float* O       = (float*)d_out;
  (void)in_sizes; (void)n_in; (void)out_size;

  const int blocks = Bsz * H * (S / 128);  // 1024
#if HAVE_TDM
  const size_t elems = (size_t)Bsz * H * S * D;   // per tensor
  const size_t need  = 2 * elems * 2;             // Kh + Vt as f16
  if (ws_size >= need) {
    _Float16* Kh = (_Float16*)d_ws;
    _Float16* Vt = Kh + elems;
    sdpa_prepack_kernel<<<Bsz * H * 8, 256, 0, stream>>>(K, V, Kh, Vt);
    sdpa_fa_tdm_kernel<<<blocks, 256, 0, stream>>>(Q, Kh, Vt, M, O);
    return;
  }
#endif
  (void)d_ws; (void)ws_size;
  sdpa_fa_stage_kernel<<<blocks, 256, 0, stream>>>(Q, K, V, M, O);
}
